// JointMMwithRel_28836410426076
// MI455X (gfx1250) — compile-verified, matrix operand order checked by reading, not run on hardware
//
#include <hip/hip_runtime.h>
#include <hip/hip_bf16.h>

// ---------------------------------------------------------------------------
// Shapes: N=16 L=128 D=768 M=8 H=8 JR=30 R=49 IMG_D=2048
// NM=128 spans, NMJR=3840 span-token rows, NR=784 image rows, HD=6144
// All GEMMs here have Ndim % 64 == 0 and Kdim % 64 == 0.
// ---------------------------------------------------------------------------

typedef __attribute__((ext_vector_type(16))) __bf16 v16bf;
typedef __attribute__((ext_vector_type(8)))  float  v8f;

union BFrag { v16bf v; unsigned short u[16]; };

__device__ __forceinline__ unsigned short f2bf(float f) {
  union { float f; unsigned int u; } c; c.f = f;
  unsigned int u = c.u;
  unsigned int rnd = 0x7FFFu + ((u >> 16) & 1u);             // RNE
  unsigned int r = (u + rnd) >> 16;
  // branchless NaN/Inf passthrough
  return (unsigned short)(((u & 0x7F800000u) == 0x7F800000u) ? (u >> 16) : r);
}
__device__ __forceinline__ void store_out(float* p, float v) { *p = v; }
__device__ __forceinline__ void store_out(unsigned short* p, float v) { *p = f2bf(v); }

// Build a 16x32 (A) or 32x16 (B) bf16 WMMA operand from an LDS tile laid out
// [16 rows][>=32 K] with row stride ldk (bf16 elems). Per ISA 7.12.2:
// lanes 0-15: K 0-7 (elems 0-7), K 16-23 (elems 8-15); lanes 16-31: K 8-15, K 24-31.
__device__ __forceinline__ v16bf make_frag(const unsigned short* base, int ldk, int lane) {
  int r    = lane & 15;
  int hsel = (lane >> 4) & 1;
  const unsigned short* p = base + r * ldk + hsel * 8;
  BFrag f;
#pragma unroll
  for (int i = 0; i < 8; ++i) { f.u[i] = p[i]; f.u[8 + i] = p[16 + i]; }
  return f.v;
}

// ---- tile loaders (per K-step of 64) --------------------------------------
// fp32 A: 64x64 tile, float4 loads, convert to bf16, b64 stores
__device__ __forceinline__ void load_tileA(const float* __restrict__ A,
                                           unsigned short (*As)[64],
                                           int tile_m, int Mdim, int Kdim,
                                           int k0, int tid) {
#pragma unroll
  for (int i = 0; i < 4; ++i) {
    int idx = tid + i * 256;              // 1024 float4 loads
    int r = idx >> 4, c4 = (idx & 15) * 4;
    int gr = tile_m + r;
    float4 v = make_float4(0.f, 0.f, 0.f, 0.f);
    if (gr < Mdim) v = *(const float4*)&A[(size_t)gr * Kdim + k0 + c4];
    unsigned int u0 = (unsigned int)f2bf(v.x) | ((unsigned int)f2bf(v.y) << 16);
    unsigned int u1 = (unsigned int)f2bf(v.z) | ((unsigned int)f2bf(v.w) << 16);
    *(uint2*)&As[r][c4] = make_uint2(u0, u1);
  }
}
// bf16 A: 64x64 tile, uint4 (8 bf16) loads, straight b128 stores
__device__ __forceinline__ void load_tileA(const unsigned short* __restrict__ A,
                                           unsigned short (*As)[64],
                                           int tile_m, int Mdim, int Kdim,
                                           int k0, int tid) {
#pragma unroll
  for (int i = 0; i < 2; ++i) {
    int idx = tid + i * 256;              // 512 uint4 loads
    int r = idx >> 3, c8 = (idx & 7) * 8;
    int gr = tile_m + r;
    uint4 v = make_uint4(0u, 0u, 0u, 0u);
    if (gr < Mdim) v = *(const uint4*)&A[(size_t)gr * Kdim + k0 + c8];
    *(uint4*)&As[r][c8] = v;
  }
}

// ---------------------------------------------------------------------------
// Generic WMMA GEMM: C[M,N] = act(A @ B + bias). 64x64 tile, K-step 64,
// 8 waves x (1 m-sub, 2 n-tiles). Requires N%64==0, K%64==0.
// ---------------------------------------------------------------------------
template <typename AT, typename OT, int ACT>
__global__ __launch_bounds__(256) void gemm_wmma(
    const AT* __restrict__ A, const float* __restrict__ B,
    const float* __restrict__ bias, OT* __restrict__ C,
    int Mdim, int Ndim, int Kdim) {
  __shared__ unsigned short As[64][64];   // [m][k]
  __shared__ unsigned short Bs[64][64];   // [n][k] (transposed in)

  const int tile_m = blockIdx.y * 64;
  const int tile_n = blockIdx.x * 64;
  const int tid = threadIdx.x, lane = tid & 31, w = tid >> 5;
  const int wm = (w & 3) * 16;
  const int wn = (w >> 2) * 32;

  v8f acc0 = {}, acc1 = {};

  for (int k0 = 0; k0 < Kdim; k0 += 64) {
    load_tileA(A, As, tile_m, Mdim, Kdim, k0, tid);
#pragma unroll
    for (int i = 0; i < 4; ++i) {         // B 64x64: float4 + transpose scatter
      int idx = tid + i * 256;
      int kk = idx >> 4, n4 = (idx & 15) * 4;
      float4 v = *(const float4*)&B[(size_t)(k0 + kk) * Ndim + tile_n + n4];
      Bs[n4 + 0][kk] = f2bf(v.x);
      Bs[n4 + 1][kk] = f2bf(v.y);
      Bs[n4 + 2][kk] = f2bf(v.z);
      Bs[n4 + 3][kk] = f2bf(v.w);
    }
    __syncthreads();
#pragma unroll
    for (int ks = 0; ks < 64; ks += 32) {
      v16bf af  = make_frag(&As[wm][ks], 64, lane);
      v16bf bf0 = make_frag(&Bs[wn][ks], 64, lane);
      v16bf bf1 = make_frag(&Bs[wn + 16][ks], 64, lane);
      acc0 = __builtin_amdgcn_wmma_f32_16x16x32_bf16(false, af, false, bf0, (short)0, acc0, false, false);
      acc1 = __builtin_amdgcn_wmma_f32_16x16x32_bf16(false, af, false, bf1, (short)0, acc1, false, false);
    }
    __syncthreads();
  }

  const int cn0 = tile_n + wn + (lane & 15);
  const int cn1 = cn0 + 16;
  const int mb  = tile_m + wm + ((lane >> 4) << 3);
#pragma unroll
  for (int r = 0; r < 8; ++r) {
    int gm = mb + r;
    if (gm >= Mdim) continue;
    float v0 = acc0[r] + (bias ? bias[cn0] : 0.f);
    float v1 = acc1[r] + (bias ? bias[cn1] : 0.f);
    if (ACT == 1) { v0 = tanhf(v0); v1 = tanhf(v1); }
    store_out(&C[(size_t)gm * Ndim + cn0], v0);
    store_out(&C[(size_t)gm * Ndim + cn1], v1);
  }
}

// ---------------------------------------------------------------------------
__global__ void offsets_kernel(const int* __restrict__ mask, int* __restrict__ woff) {
  __shared__ int len[16];
  int t = threadIdx.x;
  if (t < 16) {
    int s = 0;
    for (int l = 0; l < 128; ++l) s += mask[t * 128 + l];
    len[t] = s;
  }
  __syncthreads();
  if (t == 0) {
    int acc = 0;
    for (int n = 0; n < 16; ++n) { woff[n] = acc; acc += len[n]; }
  }
}

__global__ __launch_bounds__(256) void gather_kernel(
    const float* __restrict__ seq, const int* __restrict__ starts,
    const int* __restrict__ woff, float* __restrict__ span_emb) {
  int row = blockIdx.x;                  // sm*30 + j
  int sm = row / 30, j = row - sm * 30;
  int n = sm >> 3;
  int idx = starts[sm] + woff[n] + j;
  const int maxi = 16 * 128 - 1;
  if (idx > maxi) idx = maxi;
  const float4* src = (const float4*)(seq + (size_t)idx * 768);
  float4* dst = (float4*)(span_emb + (size_t)row * 768);
  for (int d = threadIdx.x; d < 192; d += 256) dst[d] = src[d];
}

// ---------------------------------------------------------------------------
// Fused attention per (span, head): S = q k^T / sqrt(768); softmax over 49;
// out = P V. q:[30,768] k,v:[49,768] per head, all bf16, fp32 math.
// ---------------------------------------------------------------------------
__global__ __launch_bounds__(256) void attention_kernel(
    const unsigned short* __restrict__ qw,   // [3840,6144] bf16
    const unsigned short* __restrict__ kw,   // [784,6144]  bf16
    const unsigned short* __restrict__ vw,   // [784,6144]  bf16
    unsigned short* __restrict__ attn_out) { // [3840,6144] bf16
  const int s = blockIdx.x, h = blockIdx.y;
  const int n49 = (s >> 3) * 49;
  const int tid = threadIdx.x, lane = tid & 31, w = tid >> 5;
  const int wm = (w & 1) * 16;             // 2 m sub-tiles
  const int wn = (w >> 1) * 16;            // 4 n sub-tiles
  const size_t qrow0 = (size_t)s * 30;
  const int hc = h * 768;

  __shared__ unsigned short As_[32][64];   // q tile
  __shared__ unsigned short Bs_[64][64];   // k tile / V^T tile
  __shared__ float          Sf[32][64];
  __shared__ unsigned short Pb[32][64];

  // ---- Phase 1: S = q @ k^T, K-step 64 ----
  v8f acc = {};
  for (int k0 = 0; k0 < 768; k0 += 64) {
    {                                      // q: 2048 elems = 256 uint4
      int j = tid >> 3, c8 = (tid & 7) * 8;
      uint4 v = make_uint4(0u, 0u, 0u, 0u);
      if (j < 30) v = *(const uint4*)&qw[(qrow0 + j) * 6144 + hc + k0 + c8];
      *(uint4*)&As_[j][c8] = v;
    }
#pragma unroll
    for (int i = 0; i < 2; ++i) {          // k: 4096 elems = 512 uint4
      int idx = tid + i * 256;
      int key = idx >> 3, c8 = (idx & 7) * 8;
      int kc = key < 49 ? key : 48;        // clamp addr, zero value (branchless)
      uint4 v = *(const uint4*)&kw[(size_t)(n49 + kc) * 6144 + hc + k0 + c8];
      if (key >= 49) v = make_uint4(0u, 0u, 0u, 0u);
      *(uint4*)&Bs_[key][c8] = v;
    }
    __syncthreads();
#pragma unroll
    for (int ks = 0; ks < 64; ks += 32) {
      v16bf af = make_frag(&As_[wm][ks], 64, lane);
      v16bf bf = make_frag(&Bs_[wn][ks], 64, lane);
      acc = __builtin_amdgcn_wmma_f32_16x16x32_bf16(false, af, false, bf, (short)0, acc, false, false);
    }
    __syncthreads();
  }
  {
    const float scale = 0.03608439182435161f;   // 1/sqrt(768)
    int col = wn + (lane & 15);
    int mb = wm + ((lane >> 4) << 3);
#pragma unroll
    for (int r = 0; r < 8; ++r) Sf[mb + r][col] = acc[r] * scale;
  }
  __syncthreads();

  // ---- Phase 2: masked row softmax over 49 keys ----
  if (tid < 32) {
    int rr = tid;
    float mx = -1e30f;
    if (rr < 30) for (int c = 0; c < 49; ++c) mx = fmaxf(mx, Sf[rr][c]);
    float ssum = 0.f;
    for (int c = 0; c < 64; ++c) {
      float e = (rr < 30 && c < 49) ? __expf(Sf[rr][c] - mx) : 0.f;
      Sf[rr][c] = e; ssum += e;
    }
    float inv = (ssum > 0.f) ? 1.f / ssum : 0.f;
    for (int c = 0; c < 64; ++c) Pb[rr][c] = f2bf(Sf[rr][c] * inv);
  }
  __syncthreads();

  // ---- Phase 3: out = P(32x64) @ V(64x768), K padded 49->64 with zeros ----
  for (int nc = 0; nc < 768; nc += 64) {
    v8f oacc = {};
#pragma unroll
    for (int i = 0; i < 2; ++i) {          // V chunk 64key x 64col, transpose in
      int idx = tid + i * 256;
      int key = idx >> 3, c8 = (idx & 7) * 8;
      int kc = key < 49 ? key : 48;
      uint4 v = *(const uint4*)&vw[(size_t)(n49 + kc) * 6144 + hc + nc + c8];
      if (key >= 49) v = make_uint4(0u, 0u, 0u, 0u);
      unsigned short tmp[8];
      *(uint4*)tmp = v;
#pragma unroll
      for (int e = 0; e < 8; ++e) Bs_[c8 + e][key] = tmp[e];
    }
    __syncthreads();
#pragma unroll
    for (int ks = 0; ks < 64; ks += 32) {
      v16bf af = make_frag(&Pb[wm][ks], 64, lane);
      v16bf bf = make_frag(&Bs_[wn][ks], 64, lane);
      oacc = __builtin_amdgcn_wmma_f32_16x16x32_bf16(false, af, false, bf, (short)0, oacc, false, false);
    }
    __syncthreads();                       // before next chunk rewrites Bs_
    int col = hc + nc + wn + (lane & 15);
    int mb = wm + ((lane >> 4) << 3);
#pragma unroll
    for (int r = 0; r < 8; ++r) {
      int m = mb + r;
      if (m < 30) attn_out[(qrow0 + m) * 6144 + col] = f2bf(oacc[r]);
    }
  }
}

// ---------------------------------------------------------------------------
__global__ __launch_bounds__(256) void ln_concat_kernel(
    const float* __restrict__ attfc, const float* __restrict__ span_emb,
    const float* __restrict__ g, const float* __restrict__ b,
    float* __restrict__ fused) {
  __shared__ float r1[8], r2[8];
  __shared__ float stats[2];
  int row = blockIdx.x, tid = threadIdx.x;
  const float* xa = attfc + (size_t)row * 768;
  const float* xs = span_emb + (size_t)row * 768;
  float v0 = xa[tid] + xs[tid];
  float v1 = xa[tid + 256] + xs[tid + 256];
  float v2 = xa[tid + 512] + xs[tid + 512];
  float s1 = v0 + v1 + v2;
  float s2 = v0 * v0 + v1 * v1 + v2 * v2;
  for (int off = 16; off; off >>= 1) {
    s1 += __shfl_xor(s1, off, 32);
    s2 += __shfl_xor(s2, off, 32);
  }
  int w = tid >> 5, lane = tid & 31;
  if (lane == 0) { r1[w] = s1; r2[w] = s2; }
  __syncthreads();
  if (tid == 0) {
    float a = 0.f, c = 0.f;
    for (int i = 0; i < 8; ++i) { a += r1[i]; c += r2[i]; }
    float mu = a * (1.f / 768.f);
    float var = c * (1.f / 768.f) - mu * mu;
    stats[0] = mu; stats[1] = rsqrtf(var + 1e-5f);
  }
  __syncthreads();
  float mu = stats[0], rs = stats[1];
  float* fr = fused + (size_t)row * 1536;
  fr[tid]       = xs[tid];
  fr[tid + 256] = xs[tid + 256];
  fr[tid + 512] = xs[tid + 512];
  fr[768 + tid]       = (v0 - mu) * rs * g[tid] + b[tid];
  fr[768 + tid + 256] = (v1 - mu) * rs * g[tid + 256] + b[tid + 256];
  fr[768 + tid + 512] = (v2 - mu) * rs * g[tid + 512] + b[tid + 512];
}

// ---------------------------------------------------------------------------
__global__ __launch_bounds__(256) void pool_kernel(
    const float* __restrict__ fd, const float* __restrict__ W_un,
    const float* __restrict__ b_un, const int* __restrict__ starts,
    const int* __restrict__ ends, float* __restrict__ pooled) {
  __shared__ float sc[32];
  int s = blockIdx.x;
  int tid = threadIdx.x, lane = tid & 31, w = tid >> 5;
  int width = ends[s] - starts[s] + 1;
  for (int j = w; j < 30; j += 8) {
    const float* row = fd + (size_t)(s * 30 + j) * 768;
    float acc = 0.f;
    for (int d = lane; d < 768; d += 32) acc += row[d] * W_un[d];
    for (int off = 16; off; off >>= 1) acc += __shfl_xor(acc, off, 32);
    if (lane == 0) sc[j] = acc + b_un[0];
  }
  __syncthreads();
  if (tid == 0) {
    float tmp[30];
    float mx = -1e30f;
    for (int j = 0; j < 30; ++j) {
      float v = sc[j] + (j < width ? 0.f : -10000.f);
      tmp[j] = v; if (v > mx) mx = v;
    }
    float ssum = 0.f;
    for (int j = 0; j < 30; ++j) { float e = __expf(tmp[j] - mx); tmp[j] = e; ssum += e; }
    float inv = 1.f / ssum;
    for (int j = 0; j < 30; ++j) sc[j] = tmp[j] * inv;
  }
  __syncthreads();
  for (int d = tid; d < 768; d += 256) {
    float acc = 0.f;
    for (int j = 0; j < 30; ++j) acc += sc[j] * fd[(size_t)(s * 30 + j) * 768 + d];
    pooled[(size_t)s * 768 + d] = acc;
  }
}

__global__ __launch_bounds__(256) void cls_kernel(
    const float* __restrict__ dense, const float* __restrict__ Wc,
    const float* __restrict__ bc, float* __restrict__ out) {
  int t = blockIdx.x * blockDim.x + threadIdx.x;
  if (t >= 128 * 4) return;
  int row = t >> 2, col = t & 3;
  float acc = bc[col];
  for (int d = 0; d < 768; ++d) acc += dense[row * 768 + d] * Wc[d * 4 + col];
  out[t] = acc;
}

// ---------------------------------------------------------------------------
extern "C" void kernel_launch(void* const* d_in, const int* in_sizes, int n_in,
                              void* d_out, int out_size, void* d_ws, size_t ws_size,
                              hipStream_t stream) {
  (void)in_sizes; (void)n_in; (void)out_size; (void)ws_size;
  const float* enc_img  = (const float*)d_in[0];
  const float* seq      = (const float*)d_in[1];
  const int*   mask     = (const int*)d_in[2];
  const int*   starts   = (const int*)d_in[3];
  const int*   ends     = (const int*)d_in[4];
  const float* W_align  = (const float*)d_in[5];
  const float* b_align  = (const float*)d_in[6];
  const float* Wq       = (const float*)d_in[7];
  const float* Wk       = (const float*)d_in[8];
  const float* Wv       = (const float*)d_in[9];
  const float* Wfc      = (const float*)d_in[10];
  const float* ln_g     = (const float*)d_in[11];
  const float* ln_b     = (const float*)d_in[12];
  const float* W_a1     = (const float*)d_in[13];
  const float* b_a1     = (const float*)d_in[14];
  const float* W_a2     = (const float*)d_in[15];
  const float* b_a2     = (const float*)d_in[16];
  const float* W_un     = (const float*)d_in[17];
  const float* b_un     = (const float*)d_in[18];
  const float* W_dense  = (const float*)d_in[19];
  const float* b_dense  = (const float*)d_in[20];
  const float* W_cls    = (const float*)d_in[21];
  const float* b_cls    = (const float*)d_in[22];

  char* ws = (char*)d_ws;
  size_t off = 0;
  auto carve = [&](size_t bytes) -> char* {
    char* p = ws + off;
    off += (bytes + 255) & ~(size_t)255;
    return p;
  };
  int*            woff     = (int*)carve(16 * sizeof(int));
  float*          res_feat = (float*)carve((size_t)784 * 768 * 4);
  float*          span_emb = (float*)carve((size_t)3840 * 768 * 4);
  unsigned short* kproj    = (unsigned short*)carve((size_t)784 * 6144 * 2);
  unsigned short* vproj    = (unsigned short*)carve((size_t)784 * 6144 * 2);
  char*           regionX  = carve((size_t)3840 * 6144 * 2);  // q_proj -> fused+fusedD
  char*           regionY  = carve((size_t)3840 * 6144 * 2);  // attn_out -> h1
  float*          att_fc   = (float*)carve((size_t)3840 * 768 * 4);
  float*          pooled   = (float*)carve((size_t)128 * 768 * 4);
  float*          denset   = (float*)carve((size_t)128 * 768 * 4);

  unsigned short* qproj  = (unsigned short*)regionX;
  unsigned short* attn_o = (unsigned short*)regionY;
  float*          fused  = (float*)regionX;
  float*          h1     = (float*)regionY;
  float*          fusedD = (float*)(regionX + (((size_t)3840 * 1536 * 4 + 255) & ~(size_t)255));

  dim3 blk(256);

  offsets_kernel<<<1, 32, 0, stream>>>(mask, woff);
  gather_kernel<<<3840, blk, 0, stream>>>(seq, starts, woff, span_emb);

  gemm_wmma<float, float, 0><<<dim3(768 / 64, (784 + 63) / 64), blk, 0, stream>>>(
      enc_img, W_align, b_align, res_feat, 784, 768, 2048);

  gemm_wmma<float, unsigned short, 0><<<dim3(6144 / 64, (784 + 63) / 64), blk, 0, stream>>>(
      res_feat, Wk, nullptr, kproj, 784, 6144, 768);
  gemm_wmma<float, unsigned short, 0><<<dim3(6144 / 64, (784 + 63) / 64), blk, 0, stream>>>(
      res_feat, Wv, nullptr, vproj, 784, 6144, 768);

  gemm_wmma<float, unsigned short, 0><<<dim3(6144 / 64, 3840 / 64), blk, 0, stream>>>(
      span_emb, Wq, nullptr, qproj, 3840, 6144, 768);

  attention_kernel<<<dim3(128, 8), blk, 0, stream>>>(qproj, kproj, vproj, attn_o);

  gemm_wmma<unsigned short, float, 0><<<dim3(768 / 64, 3840 / 64), blk, 0, stream>>>(
      attn_o, Wfc, nullptr, att_fc, 3840, 768, 6144);

  ln_concat_kernel<<<3840, blk, 0, stream>>>(att_fc, span_emb, ln_g, ln_b, fused);

  gemm_wmma<float, float, 1><<<dim3(1536 / 64, 3840 / 64), blk, 0, stream>>>(
      fused, W_a1, b_a1, h1, 3840, 1536, 1536);
  gemm_wmma<float, float, 0><<<dim3(768 / 64, 3840 / 64), blk, 0, stream>>>(
      h1, W_a2, b_a2, fusedD, 3840, 768, 1536);

  pool_kernel<<<128, blk, 0, stream>>>(fusedD, W_un, b_un, starts, ends, pooled);

  gemm_wmma<float, float, 1><<<dim3(768 / 64, (128 + 63) / 64), blk, 0, stream>>>(
      pooled, W_dense, b_dense, denset, 128, 768, 768);
  cls_kernel<<<2, blk, 0, stream>>>(denset, W_cls, b_cls, (float*)d_out);
}
